// SioConvLayer_55714315764410
// MI455X (gfx1250) — compile-verified, matrix operand order checked by reading, not run on hardware
//
#include <hip/hip_runtime.h>
#include <hip/hip_bf16.h>

// ---------------------------------------------------------------------------
// SioConv layer for MI455X (gfx1250, wave32, WMMA).
// - All matmul work: v_wmma_f32_16x16x32_bf16 (f32 accumulate).
// - All tile staging: global_load_async_to_lds_b128 (ASYNCcnt) so global->LDS
//   transfers overlap WMMA issue; projection GEMM is double-buffered.
// - Complex GEMMs as 4 real WMMAs; the minus term negates the imaginary A
//   fragment in registers (bf16 sign-bit XOR) - no extra LDS plane needed.
// ---------------------------------------------------------------------------

typedef __bf16 bf16;
typedef __attribute__((ext_vector_type(16))) __bf16 v16bf;
typedef __attribute__((ext_vector_type(8)))  __bf16 v8bf;
typedef __attribute__((ext_vector_type(8)))  float  v8f;
typedef __attribute__((ext_vector_type(16))) unsigned short v16us;

#define B_   4
#define L_   1024
#define DIM_ 1024
#define H_   8
#define D_   64
#define NP_  5120     // 2048 qk + 1024 v + 1024 g + 512 aa + 512 al
#define M4_  4096     // B*L

// ---- scalar helpers -------------------------------------------------------

__device__ __forceinline__ bf16 f2bf(float f) {
  unsigned u = __builtin_bit_cast(unsigned, f);
  unsigned r = (u + 0x7FFFu + ((u >> 16) & 1u)) >> 16;
  unsigned short s = (unsigned short)r;
  return __builtin_bit_cast(bf16, s);
}
__device__ __forceinline__ float bf2f(bf16 x) {
  unsigned short s = __builtin_bit_cast(unsigned short, x);
  unsigned u = ((unsigned)s) << 16;
  return __builtin_bit_cast(float, u);
}

__device__ __forceinline__ v8f wmma(v16bf a, v16bf b, v8f c) {
  return __builtin_amdgcn_wmma_f32_16x16x32_bf16(
      /*neg_a=*/false, a, /*neg_b=*/false, b,
      /*c_mod=*/(short)0, c, /*reuse_a=*/false, /*reuse_b=*/false);
}

// register-side bf16 negate of a whole fragment (sign-bit XOR)
__device__ __forceinline__ v16bf negv(v16bf a) {
  v16us x = __builtin_bit_cast(v16us, a);
  x ^= (unsigned short)0x8000u;
  return __builtin_bit_cast(v16bf, x);
}

// ---- async global -> LDS (CDNA5 Tensor/async path, ASYNCcnt tracked) ------

__device__ __forceinline__ void async_b128(void* lds_dst, const void* gsrc) {
  unsigned ldsoff = (unsigned)(size_t)lds_dst;   // generic LDS ptr: low 32b = LDS offset
  asm volatile("global_load_async_to_lds_b128 %0, %1, off"
               :: "v"(ldsoff), "v"(gsrc) : "memory");
}
__device__ __forceinline__ void async_wait0() {
  asm volatile("s_wait_asynccnt 0x0" ::: "memory");
}

// 64x64 bf16 tile copy: global (row stride rstride) -> LDS (row stride 64).
// blockDim must be 128: 2 threads per row, 32 elems (= 4x b128) each.
__device__ __forceinline__ void tile_async(bf16* dst, const bf16* src,
                                           size_t rstride, int tid) {
  int r = tid >> 1, cb = (tid & 1) * 32;
  const bf16* s = src + (size_t)r * rstride + cb;
  bf16* d = dst + r * 64 + cb;
#pragma unroll
  for (int q = 0; q < 4; ++q)
    async_b128((void*)(d + q * 8), (const void*)(s + q * 8));
}

union FragU { v16bf v; v8bf h[2]; };

// Load a 16x32 bf16 A/B fragment from a row-major LDS tile, row stride ldw.
// CDNA5 ISA 7.12.2: lane=(kh<<4)|m; elems[0..7]=K[kh*8..], [8..15]=K[16+kh*8..]
__device__ __forceinline__ v16bf ld_frag(const bf16* base, int row, int ldw,
                                         int ks, int lane) {
  const bf16* p = base + (size_t)(row + (lane & 15)) * ldw + ks + ((lane >> 4) << 3);
  FragU u;
  u.h[0] = *(const v8bf*)p;
  u.h[1] = *(const v8bf*)(p + 16);
  return u.v;
}

// Complex GEMM helper: one wave computes 16 rows x 64 cols, K=64.
// A planes (Ar,Ai): 64x64 LDS (row=out row, col=K).
// B planes (Br,Bi): 64x64 LDS (row=out col, col=K)  [Bt layout]
// Hr += Ar*Br - Ai*Bi ; Hi += Ar*Bi + Ai*Br   (minus via negv on Ai frag)
__device__ __forceinline__ void cgemm16(const bf16* Ar, const bf16* Ai, int rb,
                                        const bf16* Br, const bf16* Bi,
                                        int lane, v8f Hr[4], v8f Hi[4]) {
#pragma unroll
  for (int ks = 0; ks < 64; ks += 32) {
    v16bf ar = ld_frag(Ar, rb, 64, ks, lane);
    v16bf ai = ld_frag(Ai, rb, 64, ks, lane);
    v16bf an = negv(ai);
#pragma unroll
    for (int j = 0; j < 4; ++j) {
      v16bf br = ld_frag(Br, j * 16, 64, ks, lane);
      v16bf bi = ld_frag(Bi, j * 16, 64, ks, lane);
      Hr[j] = wmma(ar, br, Hr[j]);
      Hr[j] = wmma(an, bi, Hr[j]);
      Hi[j] = wmma(ar, bi, Hi[j]);
      Hi[j] = wmma(ai, br, Hi[j]);
    }
  }
}

// ---- conversion / layout kernels ------------------------------------------

__global__ void cvt_kernel(const float* __restrict__ in, bf16* __restrict__ out, int n) {
  int i = blockIdx.x * blockDim.x + threadIdx.x;
  if (i < n) out[i] = f2bf(in[i]);
}

// out[n*K + k] = (bf16) W[k*N + n]   (weights stored transposed for Bt tiles)
__global__ void trans_cvt(const float* __restrict__ W, bf16* __restrict__ out,
                          int K, int N) {
  int idx = blockIdx.x * blockDim.x + threadIdx.x;
  if (idx >= K * N) return;
  int n = idx / K, k = idx - n * K;
  out[idx] = f2bf(W[(size_t)k * N + n]);
}

__global__ void bcat_kernel(const float* bqk, const float* bv, const float* bg,
                            const float* baa, const float* bal, float* out) {
  int i = blockIdx.x * blockDim.x + threadIdx.x;
  if (i >= NP_) return;
  float v;
  if (i < 2048) v = bqk[i];
  else if (i < 3072) v = bv[i - 2048];
  else if (i < 4096) v = bg[i - 3072];
  else if (i < 4608) v = baa[i - 4096];
  else v = bal[i - 4608];
  out[i] = v;
}

// hidT[bh][v][i] = hidden[bh][i][v][re/im]
__global__ void hidt_kernel(const float* __restrict__ hid,
                            bf16* __restrict__ tr, bf16* __restrict__ ti) {
  int idx = blockIdx.x * blockDim.x + threadIdx.x;
  if (idx >= B_ * H_ * D_ * D_) return;
  int v = idx & 63, i = (idx >> 6) & 63, bh = idx >> 12;
  const float* p = hid + ((((size_t)bh * D_) + i) * D_ + v) * 2;
  size_t o = ((size_t)bh * D_ + v) * D_ + i;
  tr[o] = f2bf(p[0]);
  ti[o] = f2bf(p[1]);
}

// ---- generic bf16 WMMA GEMM: C[M,N] = A[M,K] * Bt[N,K]^T (+bias) ----------
// 256 threads (8 waves as 4x2), 128x128 tile, K-step 32, double-buffered
// async global->LDS staging.
__global__ void gemm_bf16(const bf16* __restrict__ A, const bf16* __restrict__ Bt,
                          const float* __restrict__ bias, float* __restrict__ C,
                          int M, int N, int K) {
  __shared__ bf16 sA[2][128 * 32];
  __shared__ bf16 sB[2][128 * 32];
  int tid = threadIdx.x, lane = tid & 31, wave = tid >> 5;
  int tn = blockIdx.x * 128, tm = blockIdx.y * 128;
  int wm = (wave >> 1) * 32, wn = (wave & 1) * 64;
  v8f acc[2][4] = {};
  int r = tid >> 1, cb = (tid & 1) * 16;
  const bf16* ga = A + (size_t)(tm + r) * K + cb;
  const bf16* gb = Bt + (size_t)(tn + r) * K + cb;
  // prologue: stage K-slice 0 into buffer 0
  async_b128(&sA[0][r * 32 + cb],     ga);
  async_b128(&sA[0][r * 32 + cb + 8], ga + 8);
  async_b128(&sB[0][r * 32 + cb],     gb);
  async_b128(&sB[0][r * 32 + cb + 8], gb + 8);
  int nk = K / 32;
  for (int k = 0; k < nk; ++k) {
    async_wait0();                 // slice k landed in LDS
    __syncthreads();               // published to all waves; prev reads done
    if (k + 1 < nk) {              // stream slice k+1 while computing slice k
      int nb = (k + 1) & 1;
      const bf16* na = ga + (size_t)(k + 1) * 32;
      const bf16* nbp = gb + (size_t)(k + 1) * 32;
      async_b128(&sA[nb][r * 32 + cb],     na);
      async_b128(&sA[nb][r * 32 + cb + 8], na + 8);
      async_b128(&sB[nb][r * 32 + cb],     nbp);
      async_b128(&sB[nb][r * 32 + cb + 8], nbp + 8);
    }
    const bf16* pa = sA[k & 1];
    const bf16* pb = sB[k & 1];
    v16bf a0 = ld_frag(pa, wm, 32, 0, lane);
    v16bf a1 = ld_frag(pa, wm + 16, 32, 0, lane);
#pragma unroll
    for (int j = 0; j < 4; ++j) {
      v16bf b = ld_frag(pb, wn + j * 16, 32, 0, lane);
      acc[0][j] = wmma(a0, b, acc[0][j]);
      acc[1][j] = wmma(a1, b, acc[1][j]);
    }
    __syncthreads();               // reads of buffer (k&1) complete
  }
  int n = lane & 15, kh = lane >> 4;
#pragma unroll
  for (int ii = 0; ii < 2; ++ii)
#pragma unroll
    for (int jj = 0; jj < 4; ++jj) {
      int col = tn + wn + jj * 16 + n;
      float bv = bias ? bias[col] : 0.f;
#pragma unroll
      for (int rr = 0; rr < 8; ++rr) {
        int row = tm + wm + ii * 16 + kh * 8 + rr;
        C[(size_t)row * N + col] = acc[ii][jj][rr] + bv;
      }
    }
}

// ---- scan: complex cumulative decay + operand plane construction ----------
// One thread per (b,h,i). Writes:
//   qa,ak,qe planes [bh][l][i]  (bf16, real+imag buffers)
//   vT planes  [bh][v=i][l]     (bf16)
//   kwT planes [bh][i][l]       (bf16, = exp(cum_m[L-1]) * ak, complex)
//   cumlast[bh*64+i] = cum at l=L-1 (complex f32)
__global__ void scan_kernel(const float* __restrict__ proj,
                            const float* __restrict__ p_angle,
                            bf16* qar, bf16* qai, bf16* akr, bf16* aki,
                            bf16* qer, bf16* qei, bf16* vTr, bf16* vTi,
                            bf16* kwTr, bf16* kwTi, float* cumlast) {
  int t = blockIdx.x * blockDim.x + threadIdx.x;
  if (t >= B_ * H_ * D_) return;
  int i = t & 63, h = (t >> 6) & 7, b = t >> 9;
  float pa_i = p_angle[h * 64 + i];
  float pa_0 = p_angle[h * 64];
  float cr = 0.f, ci = 0.f, l0r = 0.f, l0i = 0.f;
  size_t planeBase = (((size_t)b * H_ + h) * L_) * 64 + i;
  size_t vtBase = (((size_t)b * H_ + h) * 64 + i) * L_;
  for (int l = 0; l < L_; ++l) {
    const float* pr = proj + (size_t)(b * L_ + l) * NP_;
    float q_r = pr[h * 256 + i * 4 + 0], q_i = pr[h * 256 + i * 4 + 1];
    float k_r = pr[h * 256 + i * 4 + 2], k_i = pr[h * 256 + i * 4 + 3];
    float v_r = pr[2048 + h * 128 + i * 2 + 0];
    float v_i = pr[2048 + h * 128 + i * 2 + 1];
    float aa = pr[4096 + h * 64 + i];
    float al = pr[4608 + h * 64 + i];
    float sg = 1.f / (1.f + __expf(-al));
    float lr = -0.01f * sg * pa_0;
    float li = (aa + 1.f) * pa_i;
    cr += lr; ci += li;
    if (l == 0) { l0r = lr; l0i = li; }
    float cmr = cr - l0r, cmi = ci - l0i;
    float er = __expf(cmr), sn = __sinf(cmi), cs = __cosf(cmi);
    float arr = er * cs, ari = er * sn;           // a_row = exp(cum_m)
    float ei = __expf(-cmr);
    float acr = ei * cs, aci = -ei * sn;          // a_col = exp(-cum_m)
    size_t idx = planeBase + (size_t)l * 64;
    qar[idx] = f2bf(q_r * arr - q_i * ari);
    qai[idx] = f2bf(q_r * ari + q_i * arr);
    akr[idx] = f2bf(k_r * acr - k_i * aci);
    aki[idx] = f2bf(k_r * aci + k_i * acr);
    float ee = __expf(cr), s2 = __sinf(ci), c2 = __cosf(ci);
    float exr = ee * c2, exi = ee * s2;           // exp(cum)
    qer[idx] = f2bf(q_r * exr - q_i * exi);
    qei[idx] = f2bf(q_r * exi + q_i * exr);
    vTr[vtBase + l] = f2bf(v_r);
    vTi[vtBase + l] = f2bf(v_i);
  }
  cumlast[t * 2 + 0] = cr;
  cumlast[t * 2 + 1] = ci;
  // kw = exp(cum_m[L-1]) * ak   (second pass; re-reads our own bf16 writes)
  float cmr = cr - l0r, cmi = ci - l0i;
  float er = __expf(cmr);
  float eLr = er * __cosf(cmi), eLi = er * __sinf(cmi);
  for (int l = 0; l < L_; ++l) {
    size_t idx = planeBase + (size_t)l * 64;
    float ar = bf2f(akr[idx]), ai = bf2f(aki[idx]);
    kwTr[vtBase + l] = f2bf(eLr * ar - eLi * ai);
    kwTi[vtBase + l] = f2bf(eLr * ai + eLi * ar);
  }
}

// ---- causal complex attention core (flash-style over 64-row tiles) --------
__global__ void flash_kernel(const bf16* qar, const bf16* qai,
                             const bf16* akr, const bf16* aki,
                             const bf16* qer, const bf16* qei,
                             const bf16* vTr, const bf16* vTi,
                             const bf16* hidTr, const bf16* hidTi,
                             float* __restrict__ ho_r, float* __restrict__ ho_i) {
  __shared__ bf16 sQr[64 * 64], sQi[64 * 64];
  __shared__ bf16 sBr[64 * 64], sBi[64 * 64];
  __shared__ bf16 sSr[64 * 64], sSi[64 * 64];
  int I = blockIdx.x, bh = blockIdx.y;
  int tid = threadIdx.x, lane = tid & 31, wave = tid >> 5;
  size_t qbase = ((size_t)bh * L_ + (size_t)I * 64) * 64;
  v8f Hr[4] = {}, Hi[4] = {};

  // init term: H += (q*exp(cum)) @ hidden
  tile_async(sQr, qer + qbase, 64, tid);
  tile_async(sQi, qei + qbase, 64, tid);
  tile_async(sBr, hidTr + (size_t)bh * 64 * 64, 64, tid);
  tile_async(sBi, hidTi + (size_t)bh * 64 * 64, 64, tid);
  async_wait0();
  __syncthreads();
  cgemm16(sQr, sQi, wave * 16, sBr, sBi, lane, Hr, Hi);
  __syncthreads();
  tile_async(sQr, qar + qbase, 64, tid);
  tile_async(sQi, qai + qbase, 64, tid);

  int wm = (wave >> 1) * 32, wn = (wave & 1) * 32;
  for (int J = 0; J <= I; ++J) {
    size_t kb = ((size_t)bh * L_ + (size_t)J * 64) * 64;
    __syncthreads();                 // prior iteration done reading sB/sS
    tile_async(sBr, akr + kb, 64, tid);
    tile_async(sBi, aki + kb, 64, tid);
    async_wait0();                   // (also covers the qa loads on J==0)
    __syncthreads();
    // S = QA @ AK^T (complex), each wave 32x32
    v8f Sr[2][2] = {}, Si[2][2] = {};
#pragma unroll
    for (int ks = 0; ks < 64; ks += 32) {
      v16bf ar0 = ld_frag(sQr, wm, 64, ks, lane);
      v16bf ar1 = ld_frag(sQr, wm + 16, 64, ks, lane);
      v16bf ai0 = ld_frag(sQi, wm, 64, ks, lane);
      v16bf ai1 = ld_frag(sQi, wm + 16, 64, ks, lane);
      v16bf an0 = negv(ai0);
      v16bf an1 = negv(ai1);
#pragma unroll
      for (int jj = 0; jj < 2; ++jj) {
        v16bf br = ld_frag(sBr, wn + jj * 16, 64, ks, lane);
        v16bf bi = ld_frag(sBi, wn + jj * 16, 64, ks, lane);
        Sr[0][jj] = wmma(ar0, br, Sr[0][jj]);
        Sr[0][jj] = wmma(an0, bi, Sr[0][jj]);
        Si[0][jj] = wmma(ar0, bi, Si[0][jj]);
        Si[0][jj] = wmma(ai0, br, Si[0][jj]);
        Sr[1][jj] = wmma(ar1, br, Sr[1][jj]);
        Sr[1][jj] = wmma(an1, bi, Sr[1][jj]);
        Si[1][jj] = wmma(ar1, bi, Si[1][jj]);
        Si[1][jj] = wmma(ai1, br, Si[1][jj]);
      }
    }
    // causal mask (diagonal block) + C->A relayout through LDS
    {
      int n = lane & 15, kh = lane >> 4;
#pragma unroll
      for (int ii = 0; ii < 2; ++ii)
#pragma unroll
        for (int jj = 0; jj < 2; ++jj)
#pragma unroll
          for (int rr = 0; rr < 8; ++rr) {
            int row = wm + ii * 16 + kh * 8 + rr;
            int col = wn + jj * 16 + n;
            float vr = Sr[ii][jj][rr], vi = Si[ii][jj][rr];
            if (J == I && col > row) { vr = 0.f; vi = 0.f; }
            sSr[row * 64 + col] = f2bf(vr);
            sSi[row * 64 + col] = f2bf(vi);
          }
    }
    __syncthreads();                 // S tiles complete; ak reads done
    tile_async(sBr, vTr + (size_t)bh * 64 * L_ + (size_t)J * 64, L_, tid);
    tile_async(sBi, vTi + (size_t)bh * 64 * L_ + (size_t)J * 64, L_, tid);
    async_wait0();
    __syncthreads();
    cgemm16(sSr, sSi, wave * 16, sBr, sBi, lane, Hr, Hi);
  }
  // epilogue: planar f32 h [bh][l][v]
  int n = lane & 15, kh = lane >> 4;
  size_t orow0 = (size_t)bh * L_ + (size_t)I * 64 + wave * 16 + kh * 8;
#pragma unroll
  for (int j = 0; j < 4; ++j)
#pragma unroll
    for (int rr = 0; rr < 8; ++rr) {
      ho_r[(orow0 + rr) * 64 + j * 16 + n] = Hr[j][rr];
      ho_i[(orow0 + rr) * 64 + j * 16 + n] = Hi[j][rr];
    }
}

// ---- hidden_next: (64 x L) @ (L x 64) complex per (b,h) + exp(cumL)*hid ---
__global__ void hn_kernel(const bf16* kwTr, const bf16* kwTi,
                          const bf16* vTr, const bf16* vTi,
                          const float* __restrict__ cumlast,
                          const float* __restrict__ hid,
                          float* __restrict__ outH) {
  __shared__ bf16 sAr[64 * 64], sAi[64 * 64];
  __shared__ bf16 sBr[64 * 64], sBi[64 * 64];
  int bh = blockIdx.x;
  int tid = threadIdx.x, lane = tid & 31, wave = tid >> 5;
  v8f Hr[4] = {}, Hi[4] = {};
  const bf16* kr = kwTr + (size_t)bh * 64 * L_;
  const bf16* ki = kwTi + (size_t)bh * 64 * L_;
  const bf16* vr = vTr + (size_t)bh * 64 * L_;
  const bf16* vi = vTi + (size_t)bh * 64 * L_;
  for (int k0 = 0; k0 < L_; k0 += 64) {
    __syncthreads();
    tile_async(sAr, kr + k0, L_, tid);
    tile_async(sAi, ki + k0, L_, tid);
    tile_async(sBr, vr + k0, L_, tid);
    tile_async(sBi, vi + k0, L_, tid);
    async_wait0();
    __syncthreads();
    cgemm16(sAr, sAi, wave * 16, sBr, sBi, lane, Hr, Hi);
  }
  int n = lane & 15, kh = lane >> 4;
#pragma unroll
  for (int j = 0; j < 4; ++j) {
    int v = j * 16 + n;
#pragma unroll
    for (int rr = 0; rr < 8; ++rr) {
      int i = wave * 16 + kh * 8 + rr;
      float cr = cumlast[(bh * 64 + i) * 2 + 0];
      float ci = cumlast[(bh * 64 + i) * 2 + 1];
      float er = __expf(cr);
      float exr = er * __cosf(ci), exi = er * __sinf(ci);
      const float* hd = hid + (((size_t)bh * 64 + i) * 64 + v) * 2;
      float hrv = Hr[j][rr] + exr * hd[0] - exi * hd[1];
      float hiv = Hi[j][rr] + exr * hd[1] + exi * hd[0];
      float* o = outH + (((size_t)bh * 64 + i) * 64 + v) * 2;
      o[0] = hrv;
      o[1] = hiv;
    }
  }
}

// ---- GroupNorm (per b,l,head over 128 values) + SiLU gate -> bf16 acts ----
__global__ void gn_gate(const float* __restrict__ hr, const float* __restrict__ hi,
                        const float* __restrict__ proj,
                        const float* __restrict__ gn_w, const float* __restrict__ gn_b,
                        bf16* __restrict__ act) {
  int bl = blockIdx.x;                  // b*L + l
  int b = bl >> 10, l = bl & 1023;
  int wave = threadIdx.x >> 5, lane = threadIdx.x & 31;  // wave == head
  int bh = b * H_ + wave;
  const float* pr = hr + ((size_t)bh * L_ + l) * 64;
  const float* pi = hi + ((size_t)bh * L_ + l) * 64;
  float a0 = pr[lane], a1 = pi[lane], a2 = pr[lane + 32], a3 = pi[lane + 32];
  float s1 = a0 + a1 + a2 + a3;
  float s2 = a0 * a0 + a1 * a1 + a2 * a2 + a3 * a3;
  for (int off = 16; off > 0; off >>= 1) {
    s1 += __shfl_xor(s1, off);
    s2 += __shfl_xor(s2, off);
  }
  float mu = s1 * (1.f / 128.f);
  float var = s2 * (1.f / 128.f) - mu * mu;
  float rs = rsqrtf(var + 1e-5f);
  float gw = gn_w[wave], gb = gn_b[wave];
  const float* g = proj + (size_t)bl * NP_ + 3072 + wave * 128;
  bf16* o = act + (size_t)bl * DIM_ + wave * 128;
  {
    float n0 = (a0 - mu) * rs * gw + gb, n1 = (a1 - mu) * rs * gw + gb;
    float g0 = g[lane * 2], g1 = g[lane * 2 + 1];
    o[lane * 2]     = f2bf(n0 * g0 / (1.f + __expf(-g0)));
    o[lane * 2 + 1] = f2bf(n1 * g1 / (1.f + __expf(-g1)));
  }
  {
    int v = lane + 32;
    float n0 = (a2 - mu) * rs * gw + gb, n1 = (a3 - mu) * rs * gw + gb;
    float g0 = g[v * 2], g1 = g[v * 2 + 1];
    o[v * 2]     = f2bf(n0 * g0 / (1.f + __expf(-g0)));
    o[v * 2 + 1] = f2bf(n1 * g1 / (1.f + __expf(-g1)));
  }
}

// ---------------------------------------------------------------------------

extern "C" void kernel_launch(void* const* d_in, const int* in_sizes, int n_in,
                              void* d_out, int out_size, void* d_ws, size_t ws_size,
                              hipStream_t stream) {
  const float* x     = (const float*)d_in[0];
  const float* hid   = (const float*)d_in[1];
  const float* W_qk  = (const float*)d_in[2];
  const float* b_qk  = (const float*)d_in[3];
  const float* W_v   = (const float*)d_in[4];
  const float* b_v   = (const float*)d_in[5];
  const float* W_g   = (const float*)d_in[6];
  const float* b_g   = (const float*)d_in[7];
  const float* W_aa  = (const float*)d_in[8];
  const float* b_aa  = (const float*)d_in[9];
  const float* W_al  = (const float*)d_in[10];
  const float* b_al  = (const float*)d_in[11];
  const float* W_y   = (const float*)d_in[12];
  const float* b_y   = (const float*)d_in[13];
  const float* gn_w  = (const float*)d_in[14];
  const float* gn_b  = (const float*)d_in[15];
  const float* p_ang = (const float*)d_in[16];

  float* outY = (float*)d_out;
  float* outH = outY + (size_t)M4_ * DIM_;

  char* base = (char*)d_ws;
  size_t off = 0;
  auto alloc = [&](size_t bytes) -> char* {
    char* r = base + off;
    off += (bytes + 255) & ~(size_t)255;
    return r;
  };
  const size_t PLANE = (size_t)B_ * H_ * L_ * 64;      // bf16 plane elements
  bf16*  xbf   = (bf16*)alloc((size_t)M4_ * DIM_ * 2);
  bf16*  wcat  = (bf16*)alloc((size_t)NP_ * DIM_ * 2); // transposed [n][k]
  float* bcat  = (float*)alloc(NP_ * 4);
  bf16*  wyT   = (bf16*)alloc((size_t)DIM_ * DIM_ * 2);
  bf16*  hidTr = (bf16*)alloc((size_t)B_ * H_ * D_ * D_ * 2);
  bf16*  hidTi = (bf16*)alloc((size_t)B_ * H_ * D_ * D_ * 2);
  float* proj  = (float*)alloc((size_t)M4_ * NP_ * 4);
  bf16*  qar   = (bf16*)alloc(PLANE * 2);
  bf16*  qai   = (bf16*)alloc(PLANE * 2);
  bf16*  akr   = (bf16*)alloc(PLANE * 2);
  bf16*  aki   = (bf16*)alloc(PLANE * 2);
  bf16*  qer   = (bf16*)alloc(PLANE * 2);
  bf16*  qei   = (bf16*)alloc(PLANE * 2);
  bf16*  vTr   = (bf16*)alloc(PLANE * 2);
  bf16*  vTi   = (bf16*)alloc(PLANE * 2);
  bf16*  kwTr  = (bf16*)alloc(PLANE * 2);
  bf16*  kwTi  = (bf16*)alloc(PLANE * 2);
  float* cuml  = (float*)alloc((size_t)B_ * H_ * D_ * 2 * 4);
  float* hrb   = (float*)alloc(PLANE * 4);
  float* hib   = (float*)alloc(PLANE * 4);
  bf16*  act   = (bf16*)alloc((size_t)M4_ * DIM_ * 2);
  (void)ws_size; (void)in_sizes; (void)n_in; (void)out_size;

  // 1) conversions / layout
  {
    int n = M4_ * DIM_;
    cvt_kernel<<<(n + 255) / 256, 256, 0, stream>>>(x, xbf, n);
  }
  trans_cvt<<<(1024 * 2048 + 255) / 256, 256, 0, stream>>>(W_qk, wcat + (size_t)0 * DIM_, DIM_, 2048);
  trans_cvt<<<(1024 * 1024 + 255) / 256, 256, 0, stream>>>(W_v,  wcat + (size_t)2048 * DIM_, DIM_, 1024);
  trans_cvt<<<(1024 * 1024 + 255) / 256, 256, 0, stream>>>(W_g,  wcat + (size_t)3072 * DIM_, DIM_, 1024);
  trans_cvt<<<(1024 * 512  + 255) / 256, 256, 0, stream>>>(W_aa, wcat + (size_t)4096 * DIM_, DIM_, 512);
  trans_cvt<<<(1024 * 512  + 255) / 256, 256, 0, stream>>>(W_al, wcat + (size_t)4608 * DIM_, DIM_, 512);
  trans_cvt<<<(1024 * 1024 + 255) / 256, 256, 0, stream>>>(W_y, wyT, DIM_, DIM_);
  bcat_kernel<<<(NP_ + 255) / 256, 256, 0, stream>>>(b_qk, b_v, b_g, b_aa, b_al, bcat);
  hidt_kernel<<<(B_ * H_ * D_ * D_ + 255) / 256, 256, 0, stream>>>(hid, hidTr, hidTi);

  // 2) fused input projections: (4096 x 1024) x (1024 x 5120)
  gemm_bf16<<<dim3(NP_ / 128, M4_ / 128), 256, 0, stream>>>(xbf, wcat, bcat, proj,
                                                            M4_, NP_, DIM_);

  // 3) complex decay scan + operand planes
  scan_kernel<<<(B_ * H_ * D_) / 256, 256, 0, stream>>>(
      proj, p_ang, qar, qai, akr, aki, qer, qei, vTr, vTi, kwTr, kwTi, cuml);

  // 4) causal complex attention core
  flash_kernel<<<dim3(L_ / 64, B_ * H_), 128, 0, stream>>>(
      qar, qai, akr, aki, qer, qei, vTr, vTi, hidTr, hidTi, hrb, hib);

  // 5) hidden_next
  hn_kernel<<<B_ * H_, 128, 0, stream>>>(kwTr, kwTi, vTr, vTi, cuml, hid, outH);

  // 6) GroupNorm + SiLU gate
  gn_gate<<<M4_, 256, 0, stream>>>(hrb, hib, proj, gn_w, gn_b, act);

  // 7) output projection -> y
  gemm_bf16<<<dim3(DIM_ / 128, M4_ / 128), 256, 0, stream>>>(act, wyT, b_y, outY,
                                                             M4_, DIM_, DIM_);
}
